// KANLayer_7962869366953
// MI455X (gfx1250) — matrix operand order repaired and checked by reading, not compiled, not run on hardware
//
#include <hip/hip_runtime.h>

typedef __attribute__((ext_vector_type(16))) _Float16 v16h;
typedef __attribute__((ext_vector_type(8)))  _Float16 v8h;
typedef __attribute__((ext_vector_type(8)))  float    v8f;

#define UNITS   512
#define IN_DIM  512
#define BATCH   16384
#define NB      8                       // G+K basis functions per feature
#define KSPL    (IN_DIM * NB)           // 4096 spline columns
#define KDIM    (KSPL + IN_DIM)         // 4608 total K
#define NKT     (KDIM / 32)             // 144 K-chunks
#define BM      128
#define BN      128
#define BK      32
#define LDA     40                      // padded LDS row stride (halfs): 80B, 16B-aligned

// ---------------- B-spline (degree 3, 12-entry padded knot vector) ----------
__device__ __forceinline__ void bspline8(float x, const float* __restrict__ t,
                                         float* __restrict__ outv) {
    float B[11];
#pragma unroll
    for (int i = 0; i < 11; ++i)
        B[i] = (x >= t[i] && x < t[i + 1]) ? 1.0f : 0.0f;
#pragma unroll
    for (int p = 1; p <= 3; ++p) {
        const int m = 11 - p;
#pragma unroll
        for (int i = 0; i < 10; ++i) {
            if (i < m) {
                float ld = t[i + p] - t[i];
                float rd = t[i + p + 1] - t[i + 1];
                float L  = (ld > 0.0f) ? (x - t[i]) / ld : 0.0f;
                float R  = (rd > 0.0f) ? (t[i + p + 1] - x) / rd : 0.0f;
                B[i] = L * B[i] + R * B[i + 1];
            }
        }
    }
#pragma unroll
    for (int i = 0; i < 8; ++i) outv[i] = B[i];
}

// ---------------- Weight prep: Wt[n][k] (f16, K-contiguous per unit) --------
__global__ __launch_bounds__(256) void kan_prep_kernel(
    const float* __restrict__ coefs, const float* __restrict__ wfix,
    const float* __restrict__ wspl, _Float16* __restrict__ Wt) {
    int idx = blockIdx.x * 256 + threadIdx.x;
    if (idx >= UNITS * KDIM) return;
    int n = idx / KDIM;
    int k = idx - n * KDIM;
    float v;
    if (k < KSPL) {
        int j = k >> 3;
        int c = k & 7;
        v = coefs[((size_t)j * UNITS + n) * NB + c] * wspl[(size_t)j * UNITS + n];
    } else {
        int j = k - KSPL;
        v = wfix[(size_t)j * UNITS + n];
    }
    Wt[idx] = (_Float16)v;
}

// ---------------- Fused basis/silu + GEMM via WMMA (double-buffered) --------
__global__ __launch_bounds__(256) void kan_wmma_kernel(
    const float* __restrict__ inp, const float* __restrict__ knots,
    const _Float16* __restrict__ Wt, float* __restrict__ out) {
    __shared__ __align__(16) _Float16 aLds[2][BM * LDA];   // 2 x 10 KB
    __shared__ __align__(16) _Float16 bLds[2][BN * LDA];   // 2 x 10 KB

    const int tid   = threadIdx.x;
    const int lane  = tid & 31;
    const int wave  = tid >> 5;                 // 0..7
    const int nTile = blockIdx.x & 3;
    const int mTile = blockIdx.x >> 2;
    const int mBase = mTile * BM;
    const int nBase = nTile * BN;

    // padded knot vector in registers
    float t[12];
#pragma unroll
    for (int i = 0; i < 9; ++i) t[i] = knots[i];
    t[9] = t[10] = t[11] = t[8];

    v8f acc[8];
#pragma unroll
    for (int i = 0; i < 8; ++i)
#pragma unroll
        for (int e = 0; e < 8; ++e) acc[i][e] = 0.0f;

    // per-lane fragment coordinates (ISA 7.12.2 layouts, wave32)
    const int aRow  = wave * 16 + (lane & 15);
    const int aKoff = (lane < 16) ? 0 : 8;      // A: lanes16-31 hold K 8..15 / 24..31
    const int bCol  = lane & 15;
    const int bKoff = (lane < 16) ? 0 : 8;      // B: lanes16-31 hold K 8..15 / 24..31

    // ---- stage tiles for K-chunk kt into LDS buffer b ----
    auto fill = [&](int kt, int b) {
        // B tile: async global->LDS, 512 x 16B chunks, 2 per thread (ASYNCcnt)
#pragma unroll
        for (int q = 0; q < 2; ++q) {
            int c = tid + q * 256;
            int n = c >> 2;
            int o = c & 3;
            const _Float16* src =
                Wt + (size_t)(nBase + n) * KDIM + kt * BK + o * 8;
            unsigned lds = (unsigned)(uintptr_t)&bLds[b][n * LDA + o * 8];
            asm volatile("global_load_async_to_lds_b128 %0, %1, off"
                         :: "v"(lds), "v"(src) : "memory");
        }
        // A tile: computed on the fly (basis or silu columns)
        if (kt < KSPL / BK) {                   // basis: features kt*4 .. kt*4+3
#pragma unroll
            for (int pp = 0; pp < 2; ++pp) {
                int p   = tid + pp * 256;       // 512 (row,feature) pairs
                int row = p >> 2;
                int f   = p & 3;
                int j   = kt * 4 + f;
                float x = inp[(size_t)(mBase + row) * IN_DIM + j];
                float Bv[8];
                bspline8(x, t, Bv);
                v8h hv;
#pragma unroll
                for (int i = 0; i < 8; ++i) hv[i] = (_Float16)Bv[i];
                *(v8h*)&aLds[b][row * LDA + f * 8] = hv;
            }
        } else {                                // silu: 32 features per chunk
            int jbase = (kt - KSPL / BK) * BK;
#pragma unroll
            for (int pp = 0; pp < 2; ++pp) {
                int p   = tid + pp * 256;
                int row = p >> 2;
                int g   = p & 3;
                const float* xp =
                    inp + (size_t)(mBase + row) * IN_DIM + jbase + g * 8;
                v8h hv;
#pragma unroll
                for (int i = 0; i < 8; ++i) {
                    float xi = xp[i];
                    float s  = xi / (1.0f + __expf(-xi));
                    hv[i] = (_Float16)s;
                }
                *(v8h*)&aLds[b][row * LDA + g * 8] = hv;
            }
        }
    };

    // prologue: fill buffer 0
    fill(0, 0);
    asm volatile("s_wait_asynccnt 0x0" ::: "memory");
    __syncthreads();

    for (int kt = 0; kt < NKT; ++kt) {
        const int cur = kt & 1;
        // prefetch next chunk into the other buffer (overlaps with WMMAs below)
        if (kt + 1 < NKT) fill(kt + 1, cur ^ 1);

        // ---- 8 WMMAs: this wave's 16 rows x 128 cols, K += 32
        const _Float16* aL = &aLds[cur][0];
        const _Float16* bL = &bLds[cur][0];
        v16h afrag;
        ((v8h*)&afrag)[0] = *(const v8h*)&aL[aRow * LDA + aKoff];
        ((v8h*)&afrag)[1] = *(const v8h*)&aL[aRow * LDA + 16 + aKoff];
#pragma unroll
        for (int ns = 0; ns < 8; ++ns) {
            v16h bfrag;
            ((v8h*)&bfrag)[0] = *(const v8h*)&bL[(ns * 16 + bCol) * LDA + bKoff];
            ((v8h*)&bfrag)[1] = *(const v8h*)&bL[(ns * 16 + bCol) * LDA + 16 + bKoff];
            acc[ns] = __builtin_amdgcn_wmma_f32_16x16x32_f16(
                false, afrag, false, bfrag, (short)0, acc[ns], false, false);
        }

        // async B loads for kt+1 must land before next iteration reads them
        asm volatile("s_wait_asynccnt 0x0" ::: "memory");
        __syncthreads();
    }

    // ---- epilogue: C/D layout -> global (VGPR r: M=r (lanes0-15) / r+8)
    const int rowBase = mBase + wave * 16 + ((lane < 16) ? 0 : 8);
    const int nc      = lane & 15;
#pragma unroll
    for (int ns = 0; ns < 8; ++ns) {
#pragma unroll
        for (int r = 0; r < 8; ++r) {
            out[(size_t)(rowBase + r) * UNITS + nBase + ns * 16 + nc] = acc[ns][r];
        }
    }
}

// ---------------- Launcher --------------------------------------------------
extern "C" void kernel_launch(void* const* d_in, const int* in_sizes, int n_in,
                              void* d_out, int out_size, void* d_ws, size_t ws_size,
                              hipStream_t stream) {
    (void)in_sizes; (void)n_in; (void)out_size; (void)ws_size;
    const float* inp  = (const float*)d_in[0];
    const float* kn   = (const float*)d_in[1];
    const float* co   = (const float*)d_in[2];
    const float* wfix = (const float*)d_in[3];
    const float* wspl = (const float*)d_in[4];
    float*     out = (float*)d_out;
    _Float16*  Wt  = (_Float16*)d_ws;          // 512*4608 halfs = 4.72 MB

    int total = UNITS * KDIM;
    kan_prep_kernel<<<(total + 255) / 256, 256, 0, stream>>>(co, wfix, wspl, Wt);

    dim3 grid((BATCH / BM) * (UNITS / BN));    // 128 * 4 = 512 workgroups
    kan_wmma_kernel<<<grid, 256, 0, stream>>>(inp, kn, Wt, out);
}
